// GPT_80951543595421
// MI455X (gfx1250) — matrix operand order repaired and checked by reading, not compiled, not run on hardware
//
#include <hip/hip_runtime.h>
#include <math.h>

// ---------------------------------------------------------------- constants
#define Dm    1024
#define DFFm  4096
#define Tm    1024
#define Bm    8
#define Lm    8
#define Hm    16
#define HDm   64
#define BT    (Bm * Tm)        // 8192 rows of activations

// ---------------------------------------------------------------- WMMA types
typedef __attribute__((ext_vector_type(16))) __bf16 v16bf;
typedef __attribute__((ext_vector_type(8)))  float  v8f;

struct FragBF { union { v16bf v; unsigned int u[8]; }; };

__device__ inline v8f wmma_bf16(const FragBF& a, const FragBF& b, v8f c) {
  // D = A(16x32 bf16) * B(32x16 bf16) + C(16x16 f32)
  return __builtin_amdgcn_wmma_f32_16x16x32_bf16(false, a.v, false, b.v,
                                                 (short)0, c, false, false);
}

__device__ inline unsigned short f2bf(float f) {   // round-to-nearest-even
  unsigned int u = __float_as_uint(f);
  u += 0x7FFFu + ((u >> 16) & 1u);
  return (unsigned short)(u >> 16);
}

__device__ inline float gelu_exact(float x) {
  return 0.5f * x * (1.0f + erff(x * 0.7071067811865476f));
}

// CDNA5 async memory->LDS copy (32B per lane: base and base+16, IOFFSET is
// added to BOTH the global and LDS addresses per ISA 10.x). Tracked by
// ASYNCcnt; pair with s_wait_asynccnt before the workgroup barrier.
__device__ inline void async_copy32(unsigned lds_off, const void* gptr) {
  unsigned long long ga = (unsigned long long)(uintptr_t)gptr;
  asm volatile("global_load_async_to_lds_b128 %0, %1, off\n\t"
               "global_load_async_to_lds_b128 %0, %1, off offset:16"
               :: "v"(lds_off), "v"(ga) : "memory");
}
__device__ inline void wait_async0() {
  asm volatile("s_wait_asynccnt 0x0" ::: "memory");
}

// ---------------------------------------------------------------- converters
__global__ void f32_to_bf16_kernel(const float* __restrict__ in,
                                   unsigned short* __restrict__ out, size_t n) {
  size_t i = (size_t)blockIdx.x * 256 + threadIdx.x;
  if (i < n) out[i] = f2bf(in[i]);
}

// ---------------------------------------------------------------- embedding
__global__ void embed_kernel(const int* __restrict__ x,
                             const float* __restrict__ we,
                             const float* __restrict__ pe,
                             float* __restrict__ h,
                             unsigned short* __restrict__ hb) {
  size_t i = (size_t)blockIdx.x * 256 + threadIdx.x;   // BT*D total
  int d = (int)(i & (Dm - 1));
  size_t tok = i >> 10;                                // row index in [0,BT)
  int t = (int)(tok & (Tm - 1));
  int wid = x[tok];
  float v = we[(size_t)wid * Dm + d] + pe[(size_t)t * Dm + d];
  h[i] = v;
  hb[i] = f2bf(v);
}

// ---------------------------------------------------------------- layernorm
__global__ __launch_bounds__(256)
void ln_kernel(const float* __restrict__ in,
               const float* __restrict__ sc, const float* __restrict__ bi,
               float* __restrict__ outF, unsigned short* __restrict__ outB) {
  __shared__ float rs[256], rs2[256];
  int tid = threadIdx.x;
  size_t row = blockIdx.x;
  const float* xp = in + row * Dm;
  float vals[4];
  float s = 0.f, s2 = 0.f;
#pragma unroll
  for (int i = 0; i < 4; i++) {
    float v = xp[tid + i * 256];
    vals[i] = v; s += v; s2 += v * v;
  }
  rs[tid] = s; rs2[tid] = s2;
  __syncthreads();
  for (int st = 128; st > 0; st >>= 1) {
    if (tid < st) { rs[tid] += rs[tid + st]; rs2[tid] += rs2[tid + st]; }
    __syncthreads();
  }
  float mean = rs[0] * (1.0f / Dm);
  float var  = rs2[0] * (1.0f / Dm) - mean * mean;
  float rstd = rsqrtf(var + 1e-5f);
#pragma unroll
  for (int i = 0; i < 4; i++) {
    int col = tid + i * 256;
    float y = (vals[i] - mean) * rstd * sc[col] + bi[col];
    outF[row * Dm + col] = y;
    outB[row * Dm + col] = f2bf(y);
  }
}

// ---------------------------------------------------------------- WMMA GEMM
// C[M,N] = A[M,K](bf16) * Bw[K,N](bf16) + bias ; epilogue per `mode`
//   mode 0: bias only      mode 1: bias + exact GELU      mode 2: bias + resid
// Block tile 128x128, K-step 32, 8 waves as 4(M) x 2(N), wave tile 32x64.
// Double-buffered LDS; A staged via global_load_async_to_lds_b128.
#define BMt 128
#define BNt 128
#define BKt 32
#define LDA 40   // padded LDS stride (bf16) -> 16B-aligned, conflict-free frags

__global__ __launch_bounds__(256)
void gemm_kernel(const unsigned short* __restrict__ A,
                 const unsigned short* __restrict__ Bw,
                 const float* __restrict__ bias,
                 const float* __restrict__ resid,
                 float* __restrict__ outF, unsigned short* __restrict__ outB,
                 int M, int N, int K, int mode) {
  __shared__ __align__(16) unsigned short lA[2][BMt * LDA];
  __shared__ __align__(16) unsigned short lB[2][BNt * LDA];   // transposed [n][k]

  int tid  = threadIdx.x;
  int wave = tid >> 5, lane = tid & 31;
  int half = lane >> 4, l15 = lane & 15;
  int wm = wave & 3, wn = wave >> 2;          // waves: 4 (M) x 2 (N)
  int bm0 = blockIdx.y * BMt;
  int bn0 = blockIdx.x * BNt;

  v8f zero = {0.f, 0.f, 0.f, 0.f, 0.f, 0.f, 0.f, 0.f};
  v8f acc[2][4];
#pragma unroll
  for (int i = 0; i < 2; i++)
#pragma unroll
    for (int j = 0; j < 4; j++) acc[i][j] = zero;

  // cooperative load indices
  int ar  = tid >> 1;              // A: 0..127 row
  int ac  = (tid & 1) * 16;        // A: 0 / 16 col base (32B per thread)
  int bkr = tid >> 4;              // B: 0..15 k row (and +16)
  int bnc = (tid & 15) * 8;        // B: n col base, 16B per chunk

  const unsigned short* gA = A + (size_t)(bm0 + ar) * K + ac;
  const int NK = K / BKt;

  union { uint4 q; unsigned short h[8]; } b0, b1;

  // ---- prologue: stage tile 0 into buffer 0
  async_copy32((unsigned)(uintptr_t)&lA[0][ar * LDA + ac], gA);
  b0.q = *(const uint4*)(Bw + (size_t)bkr * N + bn0 + bnc);
  b1.q = *(const uint4*)(Bw + (size_t)(bkr + 16) * N + bn0 + bnc);
#pragma unroll
  for (int i = 0; i < 8; i++) {
    lB[0][(bnc + i) * LDA + bkr]      = b0.h[i];
    lB[0][(bnc + i) * LDA + bkr + 16] = b1.h[i];
  }
  wait_async0();
  __syncthreads();

  int p = 0;
  for (int kt = 0; kt < NK; kt++) {
    bool more = (kt + 1 < NK);
    int kn = (kt + 1) * BKt;
    // ---- issue next-tile loads early (A async -> LDS, B global -> regs)
    if (more) {
      async_copy32((unsigned)(uintptr_t)&lA[1 - p][ar * LDA + ac], gA + kn);
      b0.q = *(const uint4*)(Bw + (size_t)(kn + bkr) * N + bn0 + bnc);
      b1.q = *(const uint4*)(Bw + (size_t)(kn + bkr + 16) * N + bn0 + bnc);
      if (kt + 2 < NK)
        __builtin_prefetch(Bw + (size_t)(kn + BKt + bkr) * N + bn0 + bnc, 0, 3);
    }

    // ---- fragments from current buffer
    FragBF af[2], bfr[4];
#pragma unroll
    for (int i = 0; i < 2; i++) {
      int row = wm * 32 + i * 16 + l15;
#pragma unroll
      for (int v = 0; v < 8; v++) {
        int kk = (v & 3) * 2 + (v >> 2) * 16 + half * 8;   // ISA A layout
        af[i].u[v] = *(const unsigned int*)(&lA[p][row * LDA + kk]);
      }
    }
#pragma unroll
    for (int j = 0; j < 4; j++) {
      int col = wn * 64 + j * 16 + l15;
#pragma unroll
      for (int v = 0; v < 8; v++) {
        int kk = (v & 3) * 2 + (v >> 2) * 16 + half * 8;   // ISA A layout (match A)
        bfr[j].u[v] = *(const unsigned int*)(&lB[p][col * LDA + kk]);
      }
    }
#pragma unroll
    for (int i = 0; i < 2; i++)
#pragma unroll
      for (int j = 0; j < 4; j++)
        acc[i][j] = wmma_bf16(af[i], bfr[j], acc[i][j]);

    // ---- scatter next B tile into the other buffer
    if (more) {
#pragma unroll
      for (int i = 0; i < 8; i++) {
        lB[1 - p][(bnc + i) * LDA + bkr]      = b0.h[i];
        lB[1 - p][(bnc + i) * LDA + bkr + 16] = b1.h[i];
      }
    }
    wait_async0();
    __syncthreads();
    p ^= 1;
  }

  // ---- epilogue (C layout: row = r + 8*half, col = l15 within 16x16 tile)
#pragma unroll
  for (int i = 0; i < 2; i++) {
#pragma unroll
    for (int j = 0; j < 4; j++) {
      int rbase = bm0 + wm * 32 + i * 16 + half * 8;
      int coln  = bn0 + wn * 64 + j * 16 + l15;
      float bv = bias[coln];
#pragma unroll
      for (int r = 0; r < 8; r++) {
        float val = acc[i][j][r] + bv;
        size_t idx = (size_t)(rbase + r) * N + coln;
        if (mode == 1) val = gelu_exact(val);
        if (mode == 2) val += resid[idx];
        if (outF) outF[idx] = val;
        if (outB) outB[idx] = f2bf(val);
      }
    }
  }
}

// ---------------------------------------------------------------- attention
// One wave handles (b, h, 16-query tile); flash-style online softmax.
__global__ __launch_bounds__(256)
void attn_kernel(const unsigned short* __restrict__ qb,
                 const unsigned short* __restrict__ kb,
                 const unsigned short* __restrict__ vb,
                 unsigned short* __restrict__ ob) {
  __shared__ __align__(16) unsigned short lP[8][16 * LDA];  // P tile per wave

  int wave = threadIdx.x >> 5;
  int lane = threadIdx.x & 31;
  int half = lane >> 4, l15 = lane & 15;
  int gw = blockIdx.x * 8 + wave;
  int qt = gw & 63;                 // T/16 = 64 query tiles
  int hh = (gw >> 6) & 15;
  int bb = gw >> 10;
  int q0 = qt * 16;

  // Q fragments: rows = queries, K-dim = head_dim (two 32-chunks)
  FragBF qa[2];
  {
    int trow = q0 + l15;
    const unsigned short* qrow = qb + ((size_t)(bb * Tm + trow)) * Dm + hh * HDm;
#pragma unroll
    for (int c = 0; c < 2; c++)
#pragma unroll
      for (int v = 0; v < 8; v++) {
        int kk = c * 32 + (v & 3) * 2 + (v >> 2) * 16 + half * 8;
        qa[c].u[v] = *(const unsigned int*)(qrow + kk);
      }
  }

  v8f zero = {0.f, 0.f, 0.f, 0.f, 0.f, 0.f, 0.f, 0.f};
  v8f o[4]; o[0] = zero; o[1] = zero; o[2] = zero; o[3] = zero;
  float m_[8], l_[8];
#pragma unroll
  for (int r = 0; r < 8; r++) { m_[r] = -1e30f; l_[r] = 0.f; }

  int nch = (q0 >> 5) + 1;          // 32-key chunks up to the diagonal
  for (int ch = 0; ch < nch; ch++) {
    int kc = ch * 32;
    // ---- S = Q * K^T  (two 16-key tiles)
    v8f s[2]; s[0] = zero; s[1] = zero;
#pragma unroll
    for (int j = 0; j < 2; j++) {
      int key = kc + j * 16 + l15;
      const unsigned short* krow = kb + ((size_t)(bb * Tm + key)) * Dm + hh * HDm;
#pragma unroll
      for (int c = 0; c < 2; c++) {
        FragBF kf;
#pragma unroll
        for (int v = 0; v < 8; v++) {
          int hd = c * 32 + (v & 3) * 2 + (v >> 2) * 16 + half * 8;   // A-matching layout: inner-K = head_dim
          kf.u[v] = *(const unsigned int*)(krow + hd);
        }
        s[j] = wmma_bf16(qa[c], kf, s[j]);
      }
    }
    // ---- scale, causal mask, online softmax update
#pragma unroll
    for (int r = 0; r < 8; r++) {
      int rowt = q0 + r + half * 8;
      int c0 = kc + l15, c1 = kc + 16 + l15;
      float v0 = (c0 > rowt) ? -1e30f : s[0][r] * 0.125f;
      float v1 = (c1 > rowt) ? -1e30f : s[1][r] * 0.125f;
      float mx = fmaxf(v0, v1);
#pragma unroll
      for (int off = 1; off < 16; off <<= 1)
        mx = fmaxf(mx, __shfl_xor(mx, off, 32));   // stays inside 16-lane half
      float mn = fmaxf(m_[r], mx);
      float f = __expf(m_[r] - mn);
      float p0 = __expf(v0 - mn), p1 = __expf(v1 - mn);
      float rsum = p0 + p1;
#pragma unroll
      for (int off = 1; off < 16; off <<= 1)
        rsum += __shfl_xor(rsum, off, 32);
      l_[r] = l_[r] * f + rsum;
      m_[r] = mn;
      o[0][r] *= f; o[1][r] *= f; o[2][r] *= f; o[3][r] *= f;
      s[0][r] = p0; s[1][r] = p1;
    }
    // ---- repack P (C layout) -> A layout via per-wave LDS tile
#pragma unroll
    for (int r = 0; r < 8; r++) {
      lP[wave][(r + half * 8) * LDA + l15]      = f2bf(s[0][r]);
      lP[wave][(r + half * 8) * LDA + 16 + l15] = f2bf(s[1][r]);
    }
    FragBF pa;
#pragma unroll
    for (int v = 0; v < 8; v++) {
      int kk = (v & 3) * 2 + (v >> 2) * 16 + half * 8;
      pa.u[v] = *(const unsigned int*)(&lP[wave][l15 * LDA + kk]);
    }
    // ---- O += P * V   (V tile: 32 keys x 64 head_dim)
#pragma unroll
    for (int j = 0; j < 4; j++) {
      FragBF vf;
      int hd = j * 16 + l15;
#pragma unroll
      for (int v = 0; v < 8; v++) {
        int key = kc + (v & 3) * 2 + (v >> 2) * 16 + half * 8;
        const unsigned short* vp = vb + ((size_t)(bb * Tm + key)) * Dm + hh * HDm + hd;
        unsigned int lo = vp[0], hi = vp[Dm];
        vf.u[v] = lo | (hi << 16);
      }
      o[j] = wmma_bf16(pa, vf, o[j]);
    }
  }

  // ---- normalize and store bf16 output [B*T, D]
#pragma unroll
  for (int j = 0; j < 4; j++)
#pragma unroll
    for (int r = 0; r < 8; r++) {
      float val = o[j][r] / l_[r];
      int rowt = q0 + r + half * 8;
      ob[((size_t)(bb * Tm + rowt)) * Dm + hh * HDm + j * 16 + l15] = f2bf(val);
    }
}

// ---------------------------------------------------------------- launcher
extern "C" void kernel_launch(void* const* d_in, const int* in_sizes, int n_in,
                              void* d_out, int out_size, void* d_ws, size_t ws_size,
                              hipStream_t stream) {
  (void)in_sizes; (void)n_in; (void)out_size; (void)ws_size;
  const int*   x    = (const int*)  d_in[0];
  const float* we   = (const float*)d_in[1];
  const float* pe   = (const float*)d_in[2];
  const float* wq   = (const float*)d_in[3];
  const float* bq   = (const float*)d_in[4];
  const float* wk   = (const float*)d_in[5];
  const float* bk   = (const float*)d_in[6];
  const float* wv   = (const float*)d_in[7];
  const float* bv   = (const float*)d_in[8];
  const float* wo   = (const float*)d_in[9];
  const float* bo   = (const float*)d_in[10];
  const float* l1s  = (const float*)d_in[11];
  const float* l1b  = (const float*)d_in[12];
  const float* w1   = (const float*)d_in[13];
  const float* b1   = (const float*)d_in[14];
  const float* w2   = (const float*)d_in[15];
  const float* b2   = (const float*)d_in[16];
  const float* l2s  = (const float*)d_in[17];
  const float* l2b  = (const float*)d_in[18];

  // workspace carve-out
  char* ws = (char*)d_ws;
  size_t off = 0;
  auto alloc = [&](size_t bytes) -> char* {
    char* p = ws + off;
    off += (bytes + 255) & ~(size_t)255;
    return p;
  };
  unsigned short* wq_bf = (unsigned short*)alloc((size_t)Dm * Dm * 2);
  unsigned short* wk_bf = (unsigned short*)alloc((size_t)Dm * Dm * 2);
  unsigned short* wv_bf = (unsigned short*)alloc((size_t)Dm * Dm * 2);
  unsigned short* wo_bf = (unsigned short*)alloc((size_t)Dm * Dm * 2);
  unsigned short* w1_bf = (unsigned short*)alloc((size_t)Dm * DFFm * 2);
  unsigned short* w2_bf = (unsigned short*)alloc((size_t)DFFm * Dm * 2);
  float*          h     = (float*)         alloc((size_t)BT * Dm * 4);
  unsigned short* h_bf  = (unsigned short*)alloc((size_t)BT * Dm * 2);
  unsigned short* q_bf  = (unsigned short*)alloc((size_t)BT * Dm * 2);
  unsigned short* k_bf  = (unsigned short*)alloc((size_t)BT * Dm * 2);
  unsigned short* v_bf  = (unsigned short*)alloc((size_t)BT * Dm * 2);
  unsigned short* a_bf  = (unsigned short*)alloc((size_t)BT * Dm * 2);
  float*          tmp   = (float*)         alloc((size_t)BT * Dm * 4);
  unsigned short* ff_bf = (unsigned short*)alloc((size_t)BT * DFFm * 2);

  // embeddings
  embed_kernel<<<((size_t)BT * Dm) / 256, 256, 0, stream>>>(x, we, pe, h, h_bf);

  dim3 gD (Dm / BNt,   BT / BMt);   // N=1024 GEMMs: 8 x 64
  dim3 gFF(DFFm / BNt, BT / BMt);   // N=4096 GEMM: 32 x 64
  const size_t nDD = (size_t)Dm * Dm, nDF = (size_t)Dm * DFFm;

  for (int l = 0; l < Lm; l++) {
    // per-layer weight conversion f32 -> bf16
    f32_to_bf16_kernel<<<(nDD + 255) / 256, 256, 0, stream>>>(wq + (size_t)l * nDD, wq_bf, nDD);
    f32_to_bf16_kernel<<<(nDD + 255) / 256, 256, 0, stream>>>(wk + (size_t)l * nDD, wk_bf, nDD);
    f32_to_bf16_kernel<<<(nDD + 255) / 256, 256, 0, stream>>>(wv + (size_t)l * nDD, wv_bf, nDD);
    f32_to_bf16_kernel<<<(nDD + 255) / 256, 256, 0, stream>>>(wo + (size_t)l * nDD, wo_bf, nDD);
    f32_to_bf16_kernel<<<(nDF + 255) / 256, 256, 0, stream>>>(w1 + (size_t)l * nDF, w1_bf, nDF);
    f32_to_bf16_kernel<<<(nDF + 255) / 256, 256, 0, stream>>>(w2 + (size_t)l * nDF, w2_bf, nDF);

    // QKV projections (bf16 out only)
    gemm_kernel<<<gD, 256, 0, stream>>>(h_bf, wq_bf, bq + l * Dm, nullptr, nullptr, q_bf, BT, Dm, Dm, 0);
    gemm_kernel<<<gD, 256, 0, stream>>>(h_bf, wk_bf, bk + l * Dm, nullptr, nullptr, k_bf, BT, Dm, Dm, 0);
    gemm_kernel<<<gD, 256, 0, stream>>>(h_bf, wv_bf, bv + l * Dm, nullptr, nullptr, v_bf, BT, Dm, Dm, 0);

    // flash attention: B*H*(T/16) = 8192 waves -> 1024 blocks of 8 waves
    attn_kernel<<<1024, 256, 0, stream>>>(q_bf, k_bf, v_bf, a_bf);

    // output projection + residual, then LN1
    gemm_kernel<<<gD, 256, 0, stream>>>(a_bf, wo_bf, bo + l * Dm, h, tmp, nullptr, BT, Dm, Dm, 2);
    ln_kernel<<<BT, 256, 0, stream>>>(tmp, l1s + l * Dm, l1b + l * Dm, h, h_bf);

    // FFN: GELU(h @ W1 + b1) @ W2 + b2 + residual, then LN2
    gemm_kernel<<<gFF, 256, 0, stream>>>(h_bf, w1_bf, b1 + l * DFFm, nullptr, nullptr, ff_bf, BT, DFFm, Dm, 1);
    gemm_kernel<<<gD, 256, 0, stream>>>(ff_bf, w2_bf, b2 + l * Dm, h, tmp, nullptr, BT, Dm, DFFm, 2);
    ln_kernel<<<BT, 256, 0, stream>>>(tmp, l2s + l * Dm, l2b + l * Dm, h, h_bf);
  }

  hipMemcpyAsync(d_out, h, (size_t)BT * Dm * 4, hipMemcpyDeviceToDevice, stream);
}
